// AnalogyBasedEstimation_41798621724916
// MI455X (gfx1250) — matrix operand order repaired and checked by reading, not compiled, not run on hardware
//
#include <hip/hip_runtime.h>

// ---------------- problem constants ----------------
#define Fdim   128
#define Bq     2048
#define Nt     65536
#define NCHUNK 64              // train split into 64 chunks of 1024
#define CN     (Nt / NCHUNK)   // 1024
#define NLAB   10

typedef __attribute__((ext_vector_type(16))) __bf16 v16bf;
typedef __attribute__((ext_vector_type(8)))  __bf16 v8bf;
typedef __attribute__((ext_vector_type(4)))  __bf16 v4bf;
typedef __attribute__((ext_vector_type(8)))  float  v8f;
typedef __attribute__((ext_vector_type(4)))  float  v4f;

union V16 { v16bf v; v8bf h[2]; };

struct Top3 { float d0, d1, d2; int i0, i1, i2; };

// strict-< insertion: candidates arrive in ascending train-index order per
// lane, so strict < preserves "earliest index wins" for ties (top_k semantics)
__device__ __forceinline__ void fast_insert(Top3& t, float d, int i) {
    if (d < t.d2) {
        if (d < t.d1) {
            t.d2 = t.d1; t.i2 = t.i1;
            if (d < t.d0) { t.d1 = t.d0; t.i1 = t.i0; t.d0 = d; t.i0 = i; }
            else          { t.d1 = d;    t.i1 = i; }
        } else { t.d2 = d; t.i2 = i; }
    }
}

// tie-breaking insertion (lower index wins on equal distance) for merges
__device__ __forceinline__ void tb_insert(Top3& t, float d, int i) {
    bool w2 = (d < t.d2) || (d == t.d2 && i < t.i2);
    if (!w2) return;
    bool w0 = (d < t.d0) || (d == t.d0 && i < t.i0);
    bool w1 = (d < t.d1) || (d == t.d1 && i < t.i1);
    if (w0)      { t.d2 = t.d1; t.i2 = t.i1; t.d1 = t.d0; t.i1 = t.i0; t.d0 = d; t.i0 = i; }
    else if (w1) { t.d2 = t.d1; t.i2 = t.i1; t.d1 = d;    t.i1 = i; }
    else         { t.d2 = d;    t.i2 = i; }
}

// ---------------- phase 0: bf16 hi/lo split + squared-row-sums ----------------
// one wave (32 lanes) per row of 128 floats; lane handles 4 contiguous floats
__global__ void prep_rows(const float* __restrict__ src,
                          const float* __restrict__ feat,
                          __bf16* __restrict__ hi, __bf16* __restrict__ lo,
                          float* __restrict__ sums, int rows, int useFeat) {
    const int wave = threadIdx.x >> 5, lane = threadIdx.x & 31;
    const int row  = blockIdx.x * (blockDim.x >> 5) + wave;
    if (row >= rows) return;
    const int f0 = lane * 4;
    v4f v = *(const v4f*)(src + (size_t)row * Fdim + f0);
    if (useFeat) { v4f ft = *(const v4f*)(feat + f0); v = v * ft; }
    v4bf h, l;
    float s = 0.0f;
#pragma unroll
    for (int k = 0; k < 4; ++k) {
        float x  = v[k];
        __bf16 xh = (__bf16)x;
        h[k] = xh;
        l[k] = (__bf16)(x - (float)xh);   // residual for bf16x3 precision
        s += x * x;                        // exact fp32 x**2 sum
    }
    *(v4bf*)(hi + (size_t)row * Fdim + f0) = h;
    *(v4bf*)(lo + (size_t)row * Fdim + f0) = l;
#pragma unroll
    for (int m = 16; m >= 1; m >>= 1) s += __shfl_xor(s, m, 32);
    if (lane == 0) sums[row] = s;
}

// ---------------- phase 1: WMMA distance tiles + in-register top-3 ----------------
// D[m=train][n=query]: each lane's 8 C registers hold 8 train candidates for a
// single query column -> per-lane top-3, then one lane^16 shuffle merge.
// block = 256 threads = 8 waves = 128 queries sharing one 1024-row train chunk.
__global__ void knn_phase1(const __bf16* __restrict__ w_hi, const __bf16* __restrict__ w_lo,
                           const __bf16* __restrict__ x_hi, const __bf16* __restrict__ x_lo,
                           const float* __restrict__ lq_arr, const float* __restrict__ rn_arr,
                           float* __restrict__ pd, int* __restrict__ pi) {
    const int wave   = threadIdx.x >> 5, lane = threadIdx.x & 31;
    const int qb     = blockIdx.x / NCHUNK;
    const int chunk  = blockIdx.x % NCHUNK;
    const int qt     = qb * 128 + wave * 16;
    const int tstart = chunk * CN;
    const bool lowHalf = lane < 16;
    const int  nlane   = lane & 15;
    const int  q       = qt + nlane;

    // B-matrix in contiguous k order, not ISA table order (16-bit 32x16): lane holds column n = lane&15,
    // K = k0 + (lane<16 ? 0 : 16) + 0..15 contiguous -> one 32B load
    const int offB = lowHalf ? 0 : 16;
    V16 bh[4], bl[4];
    const __bf16* xb_h = x_hi + (size_t)q * Fdim;
    const __bf16* xb_l = x_lo + (size_t)q * Fdim;
#pragma unroll
    for (int k = 0; k < 4; ++k) {
        const int f = k * 32 + offB;
        bh[k].v = *(const v16bf*)(xb_h + f);
        bl[k].v = *(const v16bf*)(xb_l + f);
    }
    const float lq = lq_arr[q];

    Top3 t; t.d0 = t.d1 = t.d2 = 3.0e38f; t.i0 = t.i1 = t.i2 = 0x7FFFFFFF;

    // A-matrix in contiguous k order, same as B (16-bit 16x32): lane holds row m = lane&15,
    // elems 0..15 = K k0+offA+0..15 contiguous
    const int offA  = lowHalf ? 0 : 16;
    const int moffC = lowHalf ? 0 : 8;   // C layout: vgpr i -> row i + moffC

    for (int t0 = tstart; t0 < tstart + CN; t0 += 16) {
        const __bf16* wr_h = w_hi + (size_t)(t0 + nlane) * Fdim;
        const __bf16* wr_l = w_lo + (size_t)(t0 + nlane) * Fdim;

        // prefetch next tile's A rows (gfx1250 global_prefetch_b8) and hoist
        // the r_n loads so they resolve before the WMMA->VALU epilogue
        __builtin_prefetch(wr_h + 16 * Fdim, 0, 3);
        __builtin_prefetch(wr_l + 16 * Fdim, 0, 3);
        v4f r0 = *(const v4f*)(rn_arr + t0 + moffC);
        v4f r1 = *(const v4f*)(rn_arr + t0 + moffC + 4);

        v8f c = {};
#pragma unroll
        for (int k = 0; k < 4; ++k) {
            const int f1 = k * 32 + offA;
            V16 ah, al;
            ah.h[0] = *(const v8bf*)(wr_h + f1);
            ah.h[1] = *(const v8bf*)(wr_h + f1 + 8);
            al.h[0] = *(const v8bf*)(wr_l + f1);
            al.h[1] = *(const v8bf*)(wr_l + f1 + 8);
            // bf16x3: hi*hi + hi*lo + lo*hi, all accumulated in fp32
            c = __builtin_amdgcn_wmma_f32_16x16x32_bf16(false, ah.v, false, bh[k].v,
                                                        (short)0, c, false, false);
            c = __builtin_amdgcn_wmma_f32_16x16x32_bf16(false, ah.v, false, bl[k].v,
                                                        (short)0, c, false, false);
            c = __builtin_amdgcn_wmma_f32_16x16x32_bf16(false, al.v, false, bh[k].v,
                                                        (short)0, c, false, false);
        }
#pragma unroll
        for (int i = 0; i < 8; ++i) {
            const float rv   = (i < 4) ? r0[i] : r1[i - 4];
            // raw v_sqrt_f32 (~1 ulp): avoids the precise-sqrt refinement tail,
            // and as a TRANS op it co-executes with the VALU/WMMA stream
            const float dist = __builtin_amdgcn_sqrtf(lq + rv) - 2.0f * c[i];
            fast_insert(t, dist, t0 + moffC + i);
        }
    }

    // merge lane pair (n, n+16): both hold candidates for the same query
    Top3 o;
    o.d0 = __shfl_xor(t.d0, 16, 32); o.i0 = __shfl_xor(t.i0, 16, 32);
    o.d1 = __shfl_xor(t.d1, 16, 32); o.i1 = __shfl_xor(t.i1, 16, 32);
    o.d2 = __shfl_xor(t.d2, 16, 32); o.i2 = __shfl_xor(t.i2, 16, 32);
    tb_insert(t, o.d0, o.i0); tb_insert(t, o.d1, o.i1); tb_insert(t, o.d2, o.i2);

    if (lowHalf) {
        const size_t base = ((size_t)q * NCHUNK + chunk) * 3;
        pd[base + 0] = t.d0; pd[base + 1] = t.d1; pd[base + 2] = t.d2;
        pi[base + 0] = t.i0; pi[base + 1] = t.i1; pi[base + 2] = t.i2;
    }
}

// ---------------- phase 2: merge 64 chunk partials per query + emit outputs ----------------
__global__ void knn_phase2(const float* __restrict__ pd, const int* __restrict__ pi,
                           const int* __restrict__ labels, float* __restrict__ out) {
    const int wave = threadIdx.x >> 5, lane = threadIdx.x & 31;
    const int q    = blockIdx.x * 4 + wave;
    const int NC   = NCHUNK * 3;           // 192 candidates
    const size_t base = (size_t)q * NC;

    Top3 t; t.d0 = t.d1 = t.d2 = 3.0e38f; t.i0 = t.i1 = t.i2 = 0x7FFFFFFF;
    for (int j = lane; j < NC; j += 32) tb_insert(t, pd[base + j], pi[base + j]);

#pragma unroll
    for (int m = 16; m >= 1; m >>= 1) {
        Top3 o;
        o.d0 = __shfl_xor(t.d0, m, 32); o.i0 = __shfl_xor(t.i0, m, 32);
        o.d1 = __shfl_xor(t.d1, m, 32); o.i1 = __shfl_xor(t.i1, m, 32);
        o.d2 = __shfl_xor(t.d2, m, 32); o.i2 = __shfl_xor(t.i2, m, 32);
        tb_insert(t, o.d0, o.i0); tb_insert(t, o.d1, o.i1); tb_insert(t, o.d2, o.i2);
    }

    if (lane == 0) {
        const int l0 = labels[t.i0], l1 = labels[t.i1], l2 = labels[t.i2];
        const int cls = (l0 + l1 + l2) / 3;             // floor div (non-negative)
        // output layout: one_hot[B,10] | values[B,3] | indices[B,3] | labels[B,3]
        float* oh = out + (size_t)q * NLAB;
#pragma unroll
        for (int c = 0; c < NLAB; ++c) oh[c] = (c == cls) ? 1.0f : 0.0f;
        float* vals = out + (size_t)Bq * NLAB + (size_t)q * 3;
        vals[0] = -t.d0; vals[1] = -t.d1; vals[2] = -t.d2;  // top_k of -distance
        int* idx = (int*)out + Bq * NLAB + Bq * 3 + q * 3;
        idx[0] = t.i0; idx[1] = t.i1; idx[2] = t.i2;
        int* lab = (int*)out + Bq * NLAB + Bq * 6 + q * 3;
        lab[0] = l0; lab[1] = l1; lab[2] = l2;
    }
}

// ---------------- workspace layout (bytes) ----------------
#define OFF_WHI  ((size_t)0)
#define OFF_WLO  ((size_t)16777216)            // + N*F*2
#define OFF_XHI  ((size_t)33554432)            // + N*F*2
#define OFF_XLO  ((size_t)34078720)            // + B*F*2
#define OFF_L    ((size_t)34603008)            // + B*F*2
#define OFF_R    ((size_t)34611200)            // + B*4
#define OFF_PD   ((size_t)34873344)            // + N*4
#define OFF_PI   ((size_t)36446208)            // + B*NCHUNK*3*4

extern "C" void kernel_launch(void* const* d_in, const int* in_sizes, int n_in,
                              void* d_out, int out_size, void* d_ws, size_t ws_size,
                              hipStream_t stream) {
    const float* x      = (const float*)d_in[0];   // [B,F]
    const float* train  = (const float*)d_in[1];   // [N,F]
    const int*   labels = (const int*)d_in[2];     // [N]
    const float* feat   = (const float*)d_in[3];   // [F]

    char* ws = (char*)d_ws;
    __bf16* w_hi = (__bf16*)(ws + OFF_WHI);
    __bf16* w_lo = (__bf16*)(ws + OFF_WLO);
    __bf16* x_hi = (__bf16*)(ws + OFF_XHI);
    __bf16* x_lo = (__bf16*)(ws + OFF_XLO);
    float*  lsum = (float*)(ws + OFF_L);
    float*  rsum = (float*)(ws + OFF_R);
    float*  pd   = (float*)(ws + OFF_PD);
    int*    pi   = (int*)(ws + OFF_PI);
    float*  out  = (float*)d_out;

    // phase 0: splits + row norms (8 waves/block -> 8 rows/block)
    prep_rows<<<Bq / 8, 256, 0, stream>>>(x,     feat, x_hi, x_lo, lsum, Bq, 0);
    prep_rows<<<Nt / 8, 256, 0, stream>>>(train, feat, w_hi, w_lo, rsum, Nt, 1);

    // phase 1: 16 query-blocks x 64 train-chunks; 8 waves/block
    knn_phase1<<<(Bq / 128) * NCHUNK, 256, 0, stream>>>(w_hi, w_lo, x_hi, x_lo,
                                                        lsum, rsum, pd, pi);

    // phase 2: one wave per query
    knn_phase2<<<Bq / 4, 128, 0, stream>>>(pd, pi, labels, out);
}
